// RelationModule_50053548868058
// MI455X (gfx1250) — matrix operand order repaired and checked by reading, not compiled, not run on hardware
//
#include <hip/hip_runtime.h>

typedef __attribute__((ext_vector_type(16))) _Float16 v16h;
typedef __attribute__((ext_vector_type(8)))  float    v8f;

// ---------------------------------------------------------------------------
// K order within each 32-block: identity (contiguous order, lanes 0-15 hold
// K 0..15, lanes 16-31 hold K 16..31), matching the B operands.
// ---------------------------------------------------------------------------
__device__ __forceinline__ int permk(int k) {
  int w = k & 31;
  int p = w;
  return (k & ~31) | p;
}

// ---------------------------------------------------------------------------
// prep: corners (4096,8,3) -> obj_center (4096,3), manual_bbox (4096,27)
// ---------------------------------------------------------------------------
__global__ void prep_kernel(const float* __restrict__ corners,
                            float* __restrict__ oc, float* __restrict__ mb) {
  int idx = blockIdx.x * blockDim.x + threadIdx.x;  // 4096
  const float* cp = corners + (size_t)idx * 24;
  float mn[3] = {1e30f, 1e30f, 1e30f}, mx[3] = {-1e30f, -1e30f, -1e30f};
  float sm[3] = {0.f, 0.f, 0.f};
  for (int c = 0; c < 8; ++c)
    for (int a = 0; a < 3; ++a) {
      float v = cp[c * 3 + a];
      sm[a] += v; mn[a] = fminf(mn[a], v); mx[a] = fmaxf(mx[a], v);
    }
  float ct[3];
  for (int a = 0; a < 3; ++a) {
    oc[(size_t)idx * 3 + a] = sm[a] * 0.125f;
    ct[a] = (mn[a] + mx[a]) * 0.5f;
    mb[(size_t)idx * 27 + a] = ct[a];
  }
  for (int c = 0; c < 8; ++c)
    for (int a = 0; a < 3; ++a)
      mb[(size_t)idx * 27 + 3 + c * 3 + a] = cp[c * 3 + a] - ct[a];
}

// ---------------------------------------------------------------------------
// gather: faithful flat-index scheme of the reference
// ---------------------------------------------------------------------------
__global__ void gather_kernel(const float* __restrict__ pc,
                              const int* __restrict__ seed,
                              const int* __restrict__ agg,
                              float* __restrict__ of) {
  int idx = blockIdx.x * blockDim.x + threadIdx.x;  // 524288
  int c = idx & 127;
  int bn = idx >> 7;
  int b = bn >> 8;
  int n = bn & 255;
  int av = agg[b * 256 + n];
  int r = seed[b * 1024 + av] + b * 128;
  int f = r * 128 + c;
  int bb = f / 5120000;
  int rem = f - bb * 5120000;
  int ch = rem / 40000;
  int p = rem - ch * 40000;
  of[idx] = pc[((size_t)bb * 40000 + p) * 134 + 6 + ch];
}

// ---------------------------------------------------------------------------
// f32 -> f16 (same layout) weight conversion
// ---------------------------------------------------------------------------
__global__ void cvt_f16_kernel(const float* __restrict__ src,
                               _Float16* __restrict__ dst, int n) {
  int i = blockIdx.x * blockDim.x + threadIdx.x;
  if (i < n) dst[i] = (_Float16)src[i];
}

// ---------------------------------------------------------------------------
// q/k/v f32 -> f16 copies in WMMA-friendly layouts:
//   qh : row-major, K-permuted within each 32-block (A-fragment contiguous)
//   kh : row-major (B-fragment contiguous)
//   vt : transposed (B,H,32,256)       (B-fragment contiguous)
// ---------------------------------------------------------------------------
__global__ void cvt_qkv_kernel(const float* __restrict__ qb,
                               const float* __restrict__ kb,
                               const float* __restrict__ vb,
                               _Float16* __restrict__ qh,
                               _Float16* __restrict__ kh,
                               _Float16* __restrict__ vt) {
  int idx = blockIdx.x * blockDim.x + threadIdx.x;  // 524288
  int d = idx & 127;
  int row = idx >> 7;
  int b = row >> 8, j = row & 255;
  qh[(size_t)row * 128 + permk(d)] = (_Float16)qb[idx];
  kh[idx] = (_Float16)kb[idx];
  vt[(((size_t)b * 4 + (d >> 5)) * 32 + (d & 31)) * 256 + j] = (_Float16)vb[idx];
}

// ---------------------------------------------------------------------------
// gemm128: C(4096x128) = A(4096x128) @ W^T + bias, W pre-converted f16 [n][k].
// Weights async-staged to LDS (shared by 8 waves); A tile converted once to
// K-permuted f16 in LDS.  mode 0: C = r ; mode 1: C += alpha * r
// ---------------------------------------------------------------------------
__global__ __launch_bounds__(256) void gemm128_kernel(
    const float* __restrict__ A, const _Float16* __restrict__ WH,
    const float* __restrict__ bias, float* __restrict__ C,
    float alpha, int accumulate) {
  __shared__ __align__(32) _Float16 WHs[128 * 128];  // 32 KB
  __shared__ __align__(32) _Float16 Ash[16 * 128];   // 4 KB
  int tid = threadIdx.x;
  int mtile = blockIdx.x;

  // async-stage full f16 weight matrix into LDS (256 thr x 8 x 16B = 32 KB)
  {
    uint32_t lbase = (uint32_t)(uintptr_t)(&WHs[0]);
    const char* g = (const char*)WH;
#pragma unroll
    for (int r = 0; r < 8; ++r) {
      uint32_t off = (uint32_t)(tid * 16 + r * 4096);
      uint32_t laddr = lbase + off;
      const void* ga = (const void*)(g + off);
      asm volatile("global_load_async_to_lds_b128 %0, %1, off"
                   :: "v"(laddr), "v"(ga) : "memory");
    }
  }
  // stage A tile: 16x128 f32 -> K-permuted f16
  {
    int row = tid >> 4, c0 = (tid & 15) * 8;
    const float* ap = A + (size_t)(mtile * 16 + row) * 128 + c0;
#pragma unroll
    for (int u = 0; u < 8; ++u)
      Ash[row * 128 + permk(c0 + u)] = (_Float16)ap[u];
  }
  asm volatile("s_wait_asynccnt 0" ::: "memory");
  __syncthreads();

  int wave = tid >> 5, lane = tid & 31;
  int half = lane >> 4, lm = lane & 15;
  int n0 = wave * 16;
  v8f acc = {};
#pragma unroll
  for (int k0 = 0; k0 < 128; k0 += 32) {
    v16h af = *(const v16h*)&Ash[lm * 128 + k0 + half * 16];
    v16h bf = *(const v16h*)&WHs[(n0 + lm) * 128 + k0 + half * 16];
    acc = __builtin_amdgcn_wmma_f32_16x16x32_f16(false, af, false, bf,
                                                 (short)0, acc, false, false);
  }
  int n = n0 + lm;
#pragma unroll
  for (int c = 0; c < 8; ++c) {
    int row = mtile * 16 + c + half * 8;
    float v = acc[c] + bias[n];
    size_t o = (size_t)row * 128 + n;
    if (accumulate) C[o] += alpha * v; else C[o] = v;
  }
}

// ---------------------------------------------------------------------------
// BatchNorm (training-mode) stats over all 4096 tokens, per channel
// ---------------------------------------------------------------------------
__global__ __launch_bounds__(256) void bn_stats_kernel(
    const float* __restrict__ x, float* __restrict__ stats) {
  __shared__ float s1[256], s2[256];
  int ch = blockIdx.x;
  float a = 0.f, b = 0.f;
  for (int m = threadIdx.x; m < 4096; m += 256) {
    float v = x[(size_t)m * 128 + ch];
    a += v; b += v * v;
  }
  s1[threadIdx.x] = a; s2[threadIdx.x] = b;
  __syncthreads();
  for (int o = 128; o > 0; o >>= 1) {
    if (threadIdx.x < o) {
      s1[threadIdx.x] += s1[threadIdx.x + o];
      s2[threadIdx.x] += s2[threadIdx.x + o];
    }
    __syncthreads();
  }
  if (threadIdx.x == 0) {
    float mean = s1[0] * (1.f / 4096.f);
    stats[ch] = mean;
    stats[128 + ch] = s2[0] * (1.f / 4096.f) - mean * mean;
  }
}

__global__ void bn_prelu_kernel(float* __restrict__ x,
                                const float* __restrict__ stats,
                                const float* __restrict__ g,
                                const float* __restrict__ bb,
                                const float* __restrict__ pw) {
  int idx = blockIdx.x * blockDim.x + threadIdx.x;  // 524288
  int n = idx & 127;
  float v = (x[idx] - stats[n]) * rsqrtf(stats[128 + n] + 1e-5f) * g[n] + bb[n];
  x[idx] = v > 0.f ? v : pw[n] * v;
}

// ---------------------------------------------------------------------------
// pair_bias: per-pair MLP 4 -> 32 (relu,LN) -> 32 (relu,LN) -> 4 => dw[b,h,i,j]
// layer2 via WMMA; hidden activations in K-permuted f16 LDS, w2 f16 in LDS.
// ---------------------------------------------------------------------------
__global__ __launch_bounds__(256) void pair_bias_kernel(
    const float* __restrict__ oc,
    const float* __restrict__ w1, const float* __restrict__ b1,
    const float* __restrict__ g1, const float* __restrict__ bb1,
    const float* __restrict__ w2, const float* __restrict__ b2,
    const float* __restrict__ g2, const float* __restrict__ bb2,
    const float* __restrict__ w3, const float* __restrict__ b3,
    float* __restrict__ dw) {
  __shared__ __align__(32) _Float16 h1s[256 * 32];  // 16 KB (K-permuted)
  __shared__ __align__(32) _Float16 w2s[32 * 32];   // 2 KB
  __shared__ float h2s[256 * 32];                   // 32 KB
  int b = blockIdx.x >> 8;
  int i = blockIdx.x & 255;
  int j = threadIdx.x;
  const float* ci = oc + (size_t)(b * 256 + i) * 3;
  const float* cj = oc + (size_t)(b * 256 + j) * 3;
  float pf[4];
  pf[0] = cj[0] - ci[0]; pf[1] = cj[1] - ci[1]; pf[2] = cj[2] - ci[2];
  pf[3] = sqrtf(pf[0] * pf[0] + pf[1] * pf[1] + pf[2] * pf[2]);

  for (int t = threadIdx.x; t < 1024; t += 256) w2s[t] = (_Float16)w2[t];

  float h[32];
  float s = 0.f, sq = 0.f;
  for (int t = 0; t < 32; ++t) {
    float v = b1[t];
    for (int u = 0; u < 4; ++u) v += pf[u] * w1[t * 4 + u];
    v = v > 0.f ? v : 0.f;
    h[t] = v; s += v;
  }
  float mean = s * (1.f / 32.f);
  for (int t = 0; t < 32; ++t) { float d = h[t] - mean; sq += d * d; }
  float inv = rsqrtf(sq * (1.f / 32.f) + 1e-5f);
  for (int t = 0; t < 32; ++t)
    h1s[threadIdx.x * 32 + permk(t)] =
        (_Float16)((h[t] - mean) * inv * g1[t] + bb1[t]);
  __syncthreads();

  // layer2: (256x32) @ w2^T(32x32), 32 WMMA tiles, 4 per wave
  int wave = threadIdx.x >> 5, lane = threadIdx.x & 31;
  int half = lane >> 4, lm = lane & 15;
#pragma unroll
  for (int q = 0; q < 4; ++q) {
    int T = wave * 4 + q;
    int m0 = (T >> 1) * 16, n0 = (T & 1) * 16;
    v16h af = *(const v16h*)&h1s[(m0 + lm) * 32 + half * 16];
    v16h bf = *(const v16h*)&w2s[(n0 + lm) * 32 + half * 16];
    v8f acc = {};
    acc = __builtin_amdgcn_wmma_f32_16x16x32_f16(false, af, false, bf,
                                                 (short)0, acc, false, false);
#pragma unroll
    for (int c = 0; c < 8; ++c)
      h2s[(m0 + c + half * 8) * 32 + n0 + lm] = acc[c];
  }
  __syncthreads();

  // +bias, relu, LN, layer3 (32->4)
  s = 0.f;
  for (int t = 0; t < 32; ++t) {
    float v = h2s[threadIdx.x * 32 + t] + b2[t];
    v = v > 0.f ? v : 0.f;
    h[t] = v; s += v;
  }
  mean = s * (1.f / 32.f); sq = 0.f;
  for (int t = 0; t < 32; ++t) { float d = h[t] - mean; sq += d * d; }
  inv = rsqrtf(sq * (1.f / 32.f) + 1e-5f);
  for (int t = 0; t < 32; ++t) h[t] = (h[t] - mean) * inv * g2[t] + bb2[t];
  for (int hh = 0; hh < 4; ++hh) {
    float v = b3[hh];
    for (int t = 0; t < 32; ++t) v += h[t] * w3[hh * 32 + t];
    dw[(((size_t)b * 4 + hh) * 256 + i) * 256 + j] = v;
  }
}

// ---------------------------------------------------------------------------
// bbox_proj: features += manual_bbox(4096x27) @ bbox_w^T(128x27) + bbox_b
// ---------------------------------------------------------------------------
__global__ void bbox_proj_kernel(const float* __restrict__ mb,
                                 const float* __restrict__ w,
                                 const float* __restrict__ bias,
                                 float* __restrict__ feat) {
  int idx = blockIdx.x * blockDim.x + threadIdx.x;  // 524288
  int n = idx & 127, m = idx >> 7;
  float acc = bias[n];
  const float* mp = mb + (size_t)m * 27;
  const float* wp = w + (size_t)n * 27;
  for (int t = 0; t < 27; ++t) acc += mp[t] * wp[t];
  feat[idx] += acc;
}

// ---------------------------------------------------------------------------
// attn: one wave per (b, h, 16-query tile).  Pure-f16 operands, all WMMA
// fragments are contiguous 32B loads.  Softmax in-wave; P stored as
// K-permuted f16 directly in LDS.
// ---------------------------------------------------------------------------
__global__ __launch_bounds__(32) void attn_kernel(
    const _Float16* __restrict__ qh, const _Float16* __restrict__ kh,
    const _Float16* __restrict__ vt, const float* __restrict__ dw,
    float* __restrict__ out) {
  __shared__ float S[16 * 256];                     // 16 KB
  __shared__ __align__(32) _Float16 Ph[16 * 256];   // 8 KB
  int blk = blockIdx.x;
  int it = blk & 15;
  int h = (blk >> 4) & 3;
  int b = blk >> 6;
  int i0 = it * 16;
  int lane = threadIdx.x;
  int half = lane >> 4, lm = lane & 15;
  const float scale = 0.17677669529663687f;  // 1/sqrt(32)

  v16h qa = *(const v16h*)
      &qh[(size_t)(b * 256 + i0 + lm) * 128 + h * 32 + half * 16];
  for (int jt = 0; jt < 16; ++jt) {
    v16h bf = *(const v16h*)
        &kh[(size_t)(b * 256 + jt * 16 + lm) * 128 + h * 32 + half * 16];
    v8f acc = {};
    acc = __builtin_amdgcn_wmma_f32_16x16x32_f16(false, qa, false, bf,
                                                 (short)0, acc, false, false);
    const float* dwp = dw + (((size_t)b * 4 + h) * 256 + i0) * 256 + jt * 16;
#pragma unroll
    for (int c = 0; c < 8; ++c) {
      int r = c + half * 8;
      S[r * 256 + jt * 16 + lm] = acc[c] * scale + dwp[(size_t)r * 256 + lm];
    }
  }
  __syncthreads();

  // softmax: 2 lanes per row, combine via shfl_xor(1)
  {
    int r = lane >> 1;
    int cb = (lane & 1) * 128;
    float mx = -1e30f;
    for (int t = 0; t < 128; ++t) mx = fmaxf(mx, S[r * 256 + cb + t]);
    mx = fmaxf(mx, __shfl_xor(mx, 1));
    float sum = 0.f;
    for (int t = 0; t < 128; ++t) {
      float e = __expf(S[r * 256 + cb + t] - mx);
      S[r * 256 + cb + t] = e; sum += e;
    }
    sum += __shfl_xor(sum, 1);
    float is = 1.f / sum;
    for (int t = 0; t < 128; ++t)
      Ph[r * 256 + permk(cb + t)] = (_Float16)(S[r * 256 + cb + t] * is);
  }
  __syncthreads();

  // P(16x256) @ V(256x32)
#pragma unroll
  for (int nt = 0; nt < 2; ++nt) {
    int n0 = nt * 16;
    v8f acc = {};
#pragma unroll
    for (int k0 = 0; k0 < 256; k0 += 32) {
      v16h pa = *(const v16h*)&Ph[lm * 256 + k0 + half * 16];
      v16h bf = *(const v16h*)
          &vt[(((size_t)b * 4 + h) * 32 + n0 + lm) * 256 + k0 + half * 16];
      acc = __builtin_amdgcn_wmma_f32_16x16x32_f16(false, pa, false, bf,
                                                   (short)0, acc, false, false);
    }
#pragma unroll
    for (int c = 0; c < 8; ++c)
      out[((size_t)(b * 256 + i0 + c + half * 8) * 128) + h * 32 + n0 + lm] =
          acc[c];
  }
}

// ---------------------------------------------------------------------------
// resid_ln: features = LN(features + proj) per token (128 ch)
// ---------------------------------------------------------------------------
__global__ __launch_bounds__(128) void resid_ln_kernel(
    float* __restrict__ feat, const float* __restrict__ proj,
    const float* __restrict__ g, const float* __restrict__ bb) {
  __shared__ float s1[128], s2[128];
  int m = blockIdx.x, t = threadIdx.x;
  float v = feat[(size_t)m * 128 + t] + proj[(size_t)m * 128 + t];
  s1[t] = v; s2[t] = v * v;
  __syncthreads();
  for (int o = 64; o > 0; o >>= 1) {
    if (t < o) { s1[t] += s1[t + o]; s2[t] += s2[t + o]; }
    __syncthreads();
  }
  float mean = s1[0] * (1.f / 128.f);
  float var = s2[0] * (1.f / 128.f) - mean * mean;
  feat[(size_t)m * 128 + t] = (v - mean) * rsqrtf(var + 1e-5f) * g[t] + bb[t];
}

// ---------------------------------------------------------------------------
extern "C" void kernel_launch(void* const* d_in, const int* in_sizes, int n_in,
                              void* d_out, int out_size, void* d_ws, size_t ws_size,
                              hipStream_t stream) {
  const float* feat_in  = (const float*)d_in[0];
  const float* corners  = (const float*)d_in[1];
  const float* pc       = (const float*)d_in[2];
  const int*   seed     = (const int*)  d_in[3];
  const int*   agg      = (const int*)  d_in[4];
  const float* fc1_w    = (const float*)d_in[5];
  const float* fc1_b    = (const float*)d_in[6];
  const float* bn_g     = (const float*)d_in[7];
  const float* bn_b     = (const float*)d_in[8];
  const float* prelu_w  = (const float*)d_in[9];
  const float* fc2_w    = (const float*)d_in[10];
  const float* fc2_b    = (const float*)d_in[11];
  const float* afc_w1   = (const float*)d_in[12];
  const float* afc_b1   = (const float*)d_in[13];
  const float* ln1g     = (const float*)d_in[14];
  const float* ln1b     = (const float*)d_in[15];
  const float* afc_w2   = (const float*)d_in[16];
  const float* afc_b2   = (const float*)d_in[17];
  const float* ln2g     = (const float*)d_in[18];
  const float* ln2b     = (const float*)d_in[19];
  const float* afc_w3   = (const float*)d_in[20];
  const float* afc_b3   = (const float*)d_in[21];
  const float* wq       = (const float*)d_in[22];
  const float* bq       = (const float*)d_in[23];
  const float* wk       = (const float*)d_in[24];
  const float* bk       = (const float*)d_in[25];
  const float* wv       = (const float*)d_in[26];
  const float* bv       = (const float*)d_in[27];
  const float* wo       = (const float*)d_in[28];
  const float* bo       = (const float*)d_in[29];
  const float* ln_g     = (const float*)d_in[30];
  const float* ln_b     = (const float*)d_in[31];
  const float* bbox_w   = (const float*)d_in[32];
  const float* bbox_b   = (const float*)d_in[33];
  const float* obj_w    = (const float*)d_in[34];
  const float* obj_b    = (const float*)d_in[35];

  float* ws = (float*)d_ws;
  float* OC   = ws;                    // 4096*3
  float* MBB  = ws + 12288;            // 4096*27
  float* OF   = ws + 122880;           // 4096*128
  float* FEAT = ws + 647168;           // 4096*128
  float* TMP  = ws + 1171456;          // 4096*128  (fc1 out, then attn out)
  float* PROJ = ws + 1695744;          // 4096*128
  float* QB   = ws + 2220032;          // 4096*128
  float* KB   = ws + 2744320;          // 4096*128
  float* VB   = ws + 3268608;          // 4096*128
  float* ST   = ws + 3792896;          // 256
  float* DW   = ws + 3793152;          // 16*4*256*256 -> ends 7987456

  _Float16* H = (_Float16*)(ws + 7987456);
  _Float16* FC1H = H;                  // 16384
  _Float16* FC2H = H + 16384;
  _Float16* WQH  = H + 32768;          // 32768 (both depths)
  _Float16* WKH  = H + 65536;
  _Float16* WVH  = H + 98304;
  _Float16* WOH  = H + 131072;
  _Float16* OBJH = H + 163840;
  _Float16* QH   = H + 196608;         // 524288
  _Float16* KH   = H + 720896;
  _Float16* VTH  = H + 1245184;

  // one-time weight conversions (f32 -> f16)
  cvt_f16_kernel<<<64, 256, 0, stream>>>(fc1_w, FC1H, 16384);
  cvt_f16_kernel<<<64, 256, 0, stream>>>(fc2_w, FC2H, 16384);
  cvt_f16_kernel<<<128, 256, 0, stream>>>(wq, WQH, 32768);
  cvt_f16_kernel<<<128, 256, 0, stream>>>(wk, WKH, 32768);
  cvt_f16_kernel<<<128, 256, 0, stream>>>(wv, WVH, 32768);
  cvt_f16_kernel<<<128, 256, 0, stream>>>(wo, WOH, 32768);
  cvt_f16_kernel<<<128, 256, 0, stream>>>(obj_w, OBJH, 32768);

  prep_kernel<<<16, 256, 0, stream>>>(corners, OC, MBB);
  gather_kernel<<<4096, 128, 0, stream>>>(pc, seed, agg, OF);

  // features_concat: fc1 -> BN(train) -> PReLU -> fc2
  gemm128_kernel<<<256, 256, 0, stream>>>(feat_in, FC1H, fc1_b, TMP, 0.f, 0);
  bn_stats_kernel<<<128, 256, 0, stream>>>(TMP, ST);
  bn_prelu_kernel<<<2048, 256, 0, stream>>>(TMP, ST, bn_g, bn_b, prelu_w);
  gemm128_kernel<<<256, 256, 0, stream>>>(TMP, FC2H, fc2_b, FEAT, 0.f, 0);

  for (int i = 0; i < 2; ++i) {
    pair_bias_kernel<<<4096, 256, 0, stream>>>(
        OC, afc_w1 + i * 128, afc_b1 + i * 32, ln1g + i * 32, ln1b + i * 32,
        afc_w2 + i * 1024, afc_b2 + i * 32, ln2g + i * 32, ln2b + i * 32,
        afc_w3 + i * 128, afc_b3 + i * 4, DW);
    gemm128_kernel<<<256, 256, 0, stream>>>(OF, OBJH + i * 16384,
                                            obj_b + i * 128, FEAT, 0.1f, 1);
    bbox_proj_kernel<<<2048, 256, 0, stream>>>(MBB, bbox_w + i * 3456,
                                               bbox_b + i * 128, FEAT);
    gemm128_kernel<<<256, 256, 0, stream>>>(FEAT, WQH + i * 16384, bq + i * 128, QB, 0.f, 0);
    gemm128_kernel<<<256, 256, 0, stream>>>(FEAT, WKH + i * 16384, bk + i * 128, KB, 0.f, 0);
    gemm128_kernel<<<256, 256, 0, stream>>>(FEAT, WVH + i * 16384, bv + i * 128, VB, 0.f, 0);
    cvt_qkv_kernel<<<2048, 256, 0, stream>>>(QB, KB, VB, QH, KH, VTH);
    attn_kernel<<<1024, 32, 0, stream>>>(QH, KH, VTH, DW, TMP);
    gemm128_kernel<<<256, 256, 0, stream>>>(TMP, WOH + i * 16384, bo + i * 128, PROJ, 0.f, 0);
    resid_ln_kernel<<<4096, 128, 0, stream>>>(FEAT, PROJ, ln_g + i * 128, ln_b + i * 128);
  }

  hipMemcpyAsync(d_out, FEAT, (size_t)524288 * sizeof(float),
                 hipMemcpyDeviceToDevice, stream);
}
